// FancyConv_91027536871911
// MI455X (gfx1250) — matrix operand order repaired and checked by reading, not compiled, not run on hardware
//
#include <hip/hip_runtime.h>
#include <hip/hip_bf16.h>

// ---------------------------------------------------------------------------
// FancyConv GNN layer for gfx1250 (CDNA5, wave32, WMMA f32_16x16x32_f16,
// async global->LDS staging when available)
// ---------------------------------------------------------------------------

typedef __attribute__((ext_vector_type(16))) _Float16 v16h;
typedef __attribute__((ext_vector_type(8)))  _Float16 v8h;
typedef __attribute__((ext_vector_type(8)))  float    v8f;

#define HDIM 128
#define EMB  8
#define LN_EPS 1e-5f
#define INV_R2 (1.0f / (0.3f * 0.3f))   // GRAV=1.0

#if defined(__AMDGCN__) && __has_builtin(__builtin_amdgcn_global_load_async_to_lds_b128) && \
    __has_builtin(__builtin_amdgcn_s_wait_asynccnt)
#define USE_ASYNC_LDS 1
#else
#define USE_ASYNC_LDS 0
#endif

#if USE_ASYNC_LDS
typedef int v4i_t __attribute__((vector_size(16)));
typedef __attribute__((address_space(1))) v4i_t* gv4i_p;   // global int4*
typedef __attribute__((address_space(3))) v4i_t* lv4i_p;   // LDS int4*
#endif

// ------------------------------ helpers ------------------------------------

__device__ inline v16h ld16g(const _Float16* p) {            // 32B-aligned global
    return *(const v16h*)p;
}
__device__ inline void st16lds(_Float16* dst, v16h v) {      // 16B-aligned LDS
    v8h lo, hi;
#pragma unroll
    for (int i = 0; i < 8; i++) { lo[i] = v[i]; hi[i] = v[i + 8]; }
    *(v8h*)dst = lo;
    *(v8h*)(dst + 8) = hi;
}

// copy 16 halves (32B) global -> LDS; async path uses GLOBAL_LOAD_ASYNC_TO_LDS
__device__ inline void cp16_g2l(_Float16* ldst, const _Float16* gsrc) {
#if USE_ASYNC_LDS
    // generic LDS pointer: low 32 bits are the LDS byte offset (ISA 10.2)
    __builtin_amdgcn_global_load_async_to_lds_b128(
        (gv4i_p)(uintptr_t)gsrc,
        (lv4i_p)(unsigned int)(uintptr_t)ldst, 0, 0);
    __builtin_amdgcn_global_load_async_to_lds_b128(
        (gv4i_p)(uintptr_t)(gsrc + 8),
        (lv4i_p)(unsigned int)(uintptr_t)(ldst + 8), 0, 0);
#else
    st16lds(ldst, ld16g(gsrc));
#endif
}
__device__ inline void async_fence_lds() {
#if USE_ASYNC_LDS
    __builtin_amdgcn_s_wait_asynccnt(0);
#endif
}

// A fragment (16x32 f16): lanes 0-15 rows M=0-15 (K 0-7 / 16-23),
// lanes 16-31 rows M=0-15 (K 8-15 / 24-31). Row-major LDS tile, stride lda halves.
__device__ inline v16h loadA(const _Float16* As, int lane, int m0, int lda) {
    const _Float16* p = As + (m0 + (lane & 15)) * lda + ((lane & 16) ? 8 : 0);
    v8h lo = *(const v8h*)p;
    v8h hi = *(const v8h*)(p + 16);
    v16h r;
#pragma unroll
    for (int i = 0; i < 8; i++) { r[i] = lo[i]; r[i + 8] = hi[i]; }
    return r;
}

// B fragment (32x16 f16) from n-major (transposed) weights, stride ldb halves:
// lane n = n0 + lane%16; lanes 0-15 K 0-7 / 16-23, lanes 16-31 K 8-15 / 24-31.
__device__ inline v16h loadB(const _Float16* Bs, int lane, int n0, int ldb) {
    const _Float16* p = Bs + (n0 + (lane & 15)) * ldb + ((lane & 16) ? 8 : 0);
    v8h lo = *(const v8h*)p;
    v8h hi = *(const v8h*)(p + 16);
    v16h r;
#pragma unroll
    for (int i = 0; i < 8; i++) { r[i] = lo[i]; r[i + 8] = hi[i]; }
    return r;
}

// order-preserving float<->uint map for atomic max
__device__ inline unsigned fkey(float f) {
    unsigned u = __float_as_uint(f);
    return (u & 0x80000000u) ? ~u : (u | 0x80000000u);
}
__device__ inline float funkey(unsigned u) {
    unsigned i = (u & 0x80000000u) ? (u ^ 0x80000000u) : ~u;
    return __uint_as_float(i);
}

// ------------------------- prep / small kernels ----------------------------

// zero scatter accumulators (must run every launch; harness doesn't re-clear ws)
__global__ void init_kernel(float* __restrict__ sum, unsigned* __restrict__ maxb,
                            float* __restrict__ cnt, int n) {
    int stride = gridDim.x * blockDim.x;
    int idx = blockIdx.x * blockDim.x + threadIdx.x;
    for (int i = idx; i < n * HDIM; i += stride) { sum[i] = 0.f; maxb[i] = 0u; }
    for (int i = idx; i < n; i += stride) cnt[i] = 0.f;
}

// hf16[n][160] = [x (128), rowmean, 0-pad...]; one wave per node
__global__ void __launch_bounds__(256) hf_kernel(const float* __restrict__ x,
                                                 _Float16* __restrict__ hf16, int n) {
    int gw = (blockIdx.x * blockDim.x + threadIdx.x) >> 5;
    int lane = threadIdx.x & 31;
    if (gw >= n) return;
    const float* row = x + (size_t)gw * HDIM;
    float v[4]; float s = 0.f;
#pragma unroll
    for (int i = 0; i < 4; i++) { v[i] = row[lane + 32 * i]; s += v[i]; }
#pragma unroll
    for (int o = 16; o > 0; o >>= 1) s += __shfl_xor(s, o);
    _Float16* out = hf16 + (size_t)gw * 160;
#pragma unroll
    for (int i = 0; i < 4; i++) out[lane + 32 * i] = (_Float16)v[i];
    if (lane == 0) out[128] = (_Float16)(s * (1.f / 128.f));
    if (lane < 31) out[129 + lane] = (_Float16)0.f;
}

// generic f32 [K][Nsrc] -> f16 transposed n-major [Np][Kp], zero padded
__global__ void wprep_kernel(const float* __restrict__ W, int K, int Nsrc, int Nreal,
                             int Np, int Kp, _Float16* __restrict__ Wt) {
    int idx = blockIdx.x * blockDim.x + threadIdx.x;
    if (idx >= Np * Kp) return;
    int nn = idx / Kp, k = idx % Kp;
    float v = (nn < Nreal && k < K) ? W[(size_t)k * Nsrc + nn] : 0.f;
    Wt[idx] = (_Float16)v;
}

// conv layer0 [258][128] interleaved -> We^T (even+odd), Wo^T (odd), [128][160]
__global__ void conv0_prep_kernel(const float* __restrict__ W0,
                                  _Float16* __restrict__ WeT, _Float16* __restrict__ WoT) {
    int idx = blockIdx.x * blockDim.x + threadIdx.x;
    if (idx >= 128 * 160) return;
    int nn = idx / 160, k = idx % 160;
    float we = 0.f, wo = 0.f;
    if (k < 129) {
        float a = W0[(size_t)(2 * k) * 128 + nn];
        float b = W0[(size_t)(2 * k + 1) * 128 + nn];
        we = a + b; wo = b;
    }
    WeT[idx] = (_Float16)we;
    WoT[idx] = (_Float16)wo;
}

// ----------------------- generic 128xK WMMA GEMM ---------------------------
// C[128 rows x 128 cols] per block: 8 waves x (16 rows x 8 n-tiles).
// flags: bit0 = LayerNorm+ReLU epilogue, bit1 = f16 output (else f32)
// Out-of-range rows are CLAMPED on the gather (row m of C depends only on row m
// of A; garbage rows are masked at the store), so staging is branch-free.
#define KLD 40   // LDS stage row stride (halves), padded vs 32 for bank spread
#define CLD 132  // LDS C row stride (floats)

__global__ void __launch_bounds__(256)
gemm128_kernel(const _Float16* __restrict__ A, int Kp,
               const _Float16* __restrict__ Wt,
               const float* __restrict__ bias, const float* __restrict__ g,
               const float* __restrict__ beta, int rows, int flags,
               _Float16* __restrict__ out16, float* __restrict__ out32) {
    __shared__ __align__(16) _Float16 As[128 * KLD];
    __shared__ __align__(16) _Float16 Bs[128 * KLD];
    __shared__ float Cs[128 * CLD];
    __shared__ float sb[128], sg[128], sbe[128], smu[128], srs[128];

    int tid = threadIdx.x, lane = tid & 31, wave = tid >> 5;
    int rowBase = blockIdx.x * 128;
    int r = tid >> 1, hcol = (tid & 1) * 16;
    int grow = rowBase + r;
    if (grow > rows - 1) grow = rows - 1;
    v8f acc[8] = {};

    for (int k0 = 0; k0 < Kp; k0 += 32) {
        __syncthreads();
        // stage A tile (128 rows x 32 halves) + B tile, direct to LDS
        cp16_g2l(As + r * KLD + hcol, A + (size_t)grow * Kp + k0 + hcol);
        cp16_g2l(Bs + r * KLD + hcol, Wt + (size_t)r * Kp + k0 + hcol);
        async_fence_lds();
        __syncthreads();
        v16h a = loadA(As, lane, wave * 16, KLD);
#pragma unroll
        for (int nt = 0; nt < 8; nt++) {
            v16h b = loadB(Bs, lane, nt * 16, KLD);
            acc[nt] = __builtin_amdgcn_wmma_f32_16x16x32_f16(
                false, a, false, b, (short)0, acc[nt], false, false);
        }
    }
    __syncthreads();
    if (tid < 128) {
        sb[tid] = bias ? bias[tid] : 0.f;
        if (flags & 1) { sg[tid] = g[tid]; sbe[tid] = beta[tid]; }
    }
    __syncthreads();
#pragma unroll
    for (int nt = 0; nt < 8; nt++)
#pragma unroll
        for (int rr = 0; rr < 8; rr++) {
            int m = wave * 16 + rr + ((lane & 16) ? 8 : 0);
            int n = nt * 16 + (lane & 15);
            Cs[m * CLD + n] = acc[nt][rr] + sb[n];
        }
    __syncthreads();
    if (flags & 1) {
        if (tid < 128) {
            float s = 0.f, s2 = 0.f;
            for (int c = 0; c < 128; c++) {
                float v = Cs[tid * CLD + c];
                s += v; s2 += v * v;
            }
            float mu = s * (1.f / 128.f);
            float var = s2 * (1.f / 128.f) - mu * mu;
            smu[tid] = mu;
            srs[tid] = rsqrtf(fmaxf(var, 0.f) + LN_EPS);
        }
        __syncthreads();
    }
    for (int idx = tid; idx < 128 * 128; idx += 256) {
        int rr = idx >> 7, c = idx & 127;
        int orow = rowBase + rr;
        if (orow >= rows) continue;
        float v = Cs[rr * CLD + c];
        if (flags & 1) v = fmaxf((v - smu[rr]) * srs[rr] * sg[c] + sbe[c], 0.f);
        if (flags & 2) out16[(size_t)orow * HDIM + c] = (_Float16)v;
        else          out32[(size_t)orow * HDIM + c] = v;
    }
}

// ------------------- spatial head: [N,128] x [128,16(8)] -------------------
__global__ void __launch_bounds__(256)
gemm_emb_kernel(const _Float16* __restrict__ A, const _Float16* __restrict__ Wt,
                const float* __restrict__ bias, int rows, float* __restrict__ sf) {
    __shared__ __align__(16) _Float16 As[128 * KLD];
    __shared__ __align__(16) _Float16 Bf[16 * 128];
    __shared__ float Cs[128 * 20];
    __shared__ float sb[16];

    int tid = threadIdx.x, lane = tid & 31, wave = tid >> 5;
    int rowBase = blockIdx.x * 128;
    int r = tid >> 1, hcol = (tid & 1) * 16;
    int grow = rowBase + r;
    if (grow > rows - 1) grow = rows - 1;
    if (tid < 16) sb[tid] = (tid < EMB) ? bias[tid] : 0.f;
    // full B (16x128 halves = 2048) once: 256 threads x 8 halves
    cp16_g2l(Bf + (tid >> 1) * 16 + (tid & 1) * 8,
             Wt + (tid >> 1) * 16 + (tid & 1) * 8);
    v8f acc = {};
    for (int k0 = 0; k0 < 128; k0 += 32) {
        __syncthreads();
        cp16_g2l(As + r * KLD + hcol, A + (size_t)grow * 128 + k0 + hcol);
        async_fence_lds();
        __syncthreads();
        v16h a = loadA(As, lane, wave * 16, KLD);
        v16h b = loadB(Bf + k0, lane, 0, 128);
        acc = __builtin_amdgcn_wmma_f32_16x16x32_f16(
            false, a, false, b, (short)0, acc, false, false);
    }
    __syncthreads();
#pragma unroll
    for (int rr = 0; rr < 8; rr++) {
        int m = wave * 16 + rr + ((lane & 16) ? 8 : 0);
        int n = lane & 15;
        Cs[m * 20 + n] = acc[rr] + sb[n];
    }
    __syncthreads();
    if (tid < 128) {
        int orow = rowBase + tid;
        if (orow < rows) {
            float s2 = 0.f;
            float v[EMB];
#pragma unroll
            for (int c = 0; c < EMB; c++) { v[c] = Cs[tid * 20 + c]; s2 += v[c] * v[c]; }
            float inv = 1.f / fmaxf(sqrtf(s2), 1e-12f);
#pragma unroll
            for (int c = 0; c < EMB; c++) sf[(size_t)orow * EMB + c] = v[c] * inv;
        }
    }
}

// --------------------------- fused edge kernel -----------------------------
// 64 edges / block (128 threads = 4 waves):
//   y = P1[start] - P2[end] + b0 -> LN+ReLU -> f16 A tile
//   C = A @ W2^T + b1, scaled by w = exp(-||sf_s - sf_t||^2 / r^2)
//   scatter: atomic sum + ordered-uint atomic max into node buckets (by end)
// W2^T is async-copied into LDS and the wait only lands before the GEMM, so
// the copy overlaps the gather / LayerNorm phases.
#define ET 64
#define YLD 132
#define ALD 136

__global__ void __launch_bounds__(128)
edge_kernel(const int* __restrict__ eidx, int E_, const float* __restrict__ sf,
            const float* __restrict__ P1, const float* __restrict__ P2,
            const _Float16* __restrict__ W2t,
            const float* __restrict__ b0, const float* __restrict__ g0,
            const float* __restrict__ be0, const float* __restrict__ b1,
            float* __restrict__ sum, unsigned* __restrict__ maxb,
            float* __restrict__ cnt) {
    __shared__ float yf[ET * YLD];
    __shared__ __align__(16) _Float16 As[ET * ALD];
    __shared__ __align__(16) _Float16 Bs[128 * ALD];
    __shared__ float wbuf[ET], smu[ET], srs[ET];
    __shared__ int startb[ET], endb[ET];
    __shared__ float sb0[128], sg0[128], sbe0[128], sb1[128];

    int tid = threadIdx.x, lane = tid & 31, wave = tid >> 5;
    int base = blockIdx.x * ET;
    const int* startArr = eidx;
    const int* endArr = eidx + E_;

    sb0[tid] = b0[tid]; sg0[tid] = g0[tid]; sbe0[tid] = be0[tid]; sb1[tid] = b1[tid];
    // W2^T (128x128 halves) async copy into padded LDS (overlaps phases below)
#pragma unroll
    for (int it = 0; it < 8; it++) {
        int i = it * 128 + tid;
        int row = i >> 3, ch = i & 7;
        cp16_g2l(Bs + row * ALD + ch * 16, W2t + row * 128 + ch * 16);
    }
    if (tid < ET) {
        int e = base + tid;
        if (e < E_) {
            int s = startArr[e], t = endArr[e];
            startb[tid] = s; endb[tid] = t;
            float d = 0.f;
#pragma unroll
            for (int j = 0; j < EMB; j++) {
                float df = sf[(size_t)s * EMB + j] - sf[(size_t)t * EMB + j];
                d += df * df;
            }
            wbuf[tid] = __expf(-d * INV_R2);
            atomicAdd(&cnt[t], 1.0f);
        } else {
            startb[tid] = -1; endb[tid] = -1; wbuf[tid] = 0.f;
        }
    }
    __syncthreads();
    // y = P1[s] - P2[t] + b0 (coalesced: 128 threads sweep one edge row / iter)
    for (int k = 0; k < ET; k++) {
        int s = startb[k], t = endb[k];
        float v = 0.f;
        if (s >= 0)
            v = P1[(size_t)s * HDIM + tid] - P2[(size_t)t * HDIM + tid] + sb0[tid];
        yf[k * YLD + tid] = v;
    }
    __syncthreads();
    if (tid < ET) {
        float s = 0.f, s2 = 0.f;
        for (int c = 0; c < 128; c++) {
            float v = yf[tid * YLD + c];
            s += v; s2 += v * v;
        }
        float mu = s * (1.f / 128.f);
        float var = s2 * (1.f / 128.f) - mu * mu;
        smu[tid] = mu;
        srs[tid] = rsqrtf(fmaxf(var, 0.f) + LN_EPS);
    }
    __syncthreads();
    for (int idx = tid; idx < ET * 128; idx += 128) {
        int e2 = idx >> 7, c = idx & 127;
        float v = fmaxf((yf[e2 * YLD + c] - smu[e2]) * srs[e2] * sg0[c] + sbe0[c], 0.f);
        As[e2 * ALD + c] = (_Float16)v;
    }
    async_fence_lds();   // W2^T copy must have landed before GEMM reads Bs
    __syncthreads();
    v8f acc[8] = {};
#pragma unroll
    for (int k0 = 0; k0 < 128; k0 += 32) {
        v16h a = loadA(As + k0, lane, wave * 16, ALD);
#pragma unroll
        for (int nt = 0; nt < 8; nt++) {
            v16h b = loadB(Bs + k0, lane, nt * 16, ALD);
            acc[nt] = __builtin_amdgcn_wmma_f32_16x16x32_f16(
                false, a, false, b, (short)0, acc[nt], false, false);
        }
    }
    // scatter (C layout: VGPR r -> M=r(+8 hi half-wave), lane%16 -> N in n-tile)
#pragma unroll
    for (int nt = 0; nt < 8; nt++)
#pragma unroll
        for (int rr = 0; rr < 8; rr++) {
            int m = wave * 16 + rr + ((lane & 16) ? 8 : 0);
            int t = endb[m];
            if (t >= 0) {
                int n = nt * 16 + (lane & 15);
                float val = (acc[nt][rr] + sb1[n]) * wbuf[m];
                atomicAdd(&sum[(size_t)t * HDIM + n], val);
                atomicMax(&maxb[(size_t)t * HDIM + n], fkey(val));
            }
        }
}

// ------------------ aggregate finalize -> cat16 [N,544] --------------------
__global__ void finalize_kernel(const float* __restrict__ sum,
                                const unsigned* __restrict__ maxb,
                                const float* __restrict__ cnt,
                                const _Float16* __restrict__ hf16,
                                _Float16* __restrict__ cat16, int n) {
    int stride = gridDim.x * blockDim.x;
    int idx = blockIdx.x * blockDim.x + threadIdx.x;
    for (int i = idx; i < n * HDIM; i += stride) {
        int nd = i >> 7, c = i & 127;
        float cv = cnt[nd];
        float s = sum[i];
        float mx = (cv > 0.f) ? funkey(maxb[i]) : 0.f;
        float mean = s / fmaxf(cv, 1.f);
        size_t b = (size_t)nd * 544 + c;
        cat16[b] = (_Float16)s;
        cat16[b + 128] = (_Float16)mx;
        cat16[b + 256] = (_Float16)mean;
    }
    for (int i = idx; i < n * 160; i += stride) {
        int nd = i / 160, j = i % 160;
        _Float16 v = (j < 129) ? hf16[(size_t)nd * 160 + j] : (_Float16)0.f;
        cat16[(size_t)nd * 544 + 384 + j] = v;   // fills cols 384..543
    }
}

// ------------------------------- host side ---------------------------------

static inline size_t alignUp(size_t v, size_t a) { return (v + a - 1) & ~(a - 1); }

extern "C" void kernel_launch(void* const* d_in, const int* in_sizes, int n_in,
                              void* d_out, int out_size, void* d_ws, size_t ws_size,
                              hipStream_t stream) {
    // ---- locate inputs by unique flat sizes (robust to flattening order) ----
    int iConv = -1, iFeat = -1, iSpat = -1, iHF = -1, iEI = -1;
    long big0 = -1, big1 = -1;
    for (int i = 0; i < n_in; i++) {
        long s = in_sizes[i];
        if (s == 258 * 128) iConv = i;
        else if (s == 513 * 128) iFeat = i;
        else if (s == 129 * 128) iSpat = i;
        if (s > big0) { big1 = big0; iEI = iHF; big0 = s; iHF = i; }
        else if (s > big1) { big1 = s; iEI = i; }
    }
    const int N = (int)(in_sizes[iHF] / HDIM);
    const int E = (int)(in_sizes[iEI] / 2);

    const float* x    = (const float*)d_in[iHF];
    const int*   eidx = (const int*)d_in[iEI];
    // conv: W0,b0,beta0,g0,W1,b1
    const float* cW0 = (const float*)d_in[iConv];
    const float* cb0 = (const float*)d_in[iConv + 1];
    const float* cbe0 = (const float*)d_in[iConv + 2];
    const float* cg0 = (const float*)d_in[iConv + 3];
    const float* cW1 = (const float*)d_in[iConv + 4];
    const float* cb1 = (const float*)d_in[iConv + 5];
    // feature: W0,b0,beta0,g0,W1,b1,beta1,g1
    const float* fW0 = (const float*)d_in[iFeat];
    const float* fb0 = (const float*)d_in[iFeat + 1];
    const float* fbe0 = (const float*)d_in[iFeat + 2];
    const float* fg0 = (const float*)d_in[iFeat + 3];
    const float* fW1 = (const float*)d_in[iFeat + 4];
    const float* fb1 = (const float*)d_in[iFeat + 5];
    const float* fbe1 = (const float*)d_in[iFeat + 6];
    const float* fg1 = (const float*)d_in[iFeat + 7];
    // spatial: W0,b0,beta0,g0,W1,b1,beta1,g1,W2,b2
    const float* sW0 = (const float*)d_in[iSpat];
    const float* sb0 = (const float*)d_in[iSpat + 1];
    const float* sbe0 = (const float*)d_in[iSpat + 2];
    const float* sg0 = (const float*)d_in[iSpat + 3];
    const float* sW1 = (const float*)d_in[iSpat + 4];
    const float* sb1 = (const float*)d_in[iSpat + 5];
    const float* sbe1 = (const float*)d_in[iSpat + 6];
    const float* sg1 = (const float*)d_in[iSpat + 7];
    const float* sW2 = (const float*)d_in[iSpat + 8];
    const float* sb2 = (const float*)d_in[iSpat + 9];

    // ---- workspace bump allocator ----
    char* ws = (char*)d_ws;
    size_t off = 0;
    auto alloc = [&](size_t bytes) -> char* {
        char* p = ws + off;
        off = alignUp(off + bytes, 256);
        return p;
    };
    _Float16* hf16 = (_Float16*)alloc((size_t)N * 160 * 2);
    _Float16* t1   = (_Float16*)alloc((size_t)N * HDIM * 2);   // reused as f1
    float*    sumB = (float*)alloc((size_t)N * HDIM * 4);
    unsigned* maxB = (unsigned*)alloc((size_t)N * HDIM * 4);
    float*    cntB = (float*)alloc((size_t)N * 4);
    float*    sfB  = (float*)alloc((size_t)N * EMB * 4);
    // weights (f16 transposed, padded)
    _Float16* W0sT = (_Float16*)alloc(128 * 160 * 2);
    _Float16* W1sT = (_Float16*)alloc(128 * 128 * 2);
    _Float16* W2sT = (_Float16*)alloc(16 * 128 * 2);
    _Float16* WeT  = (_Float16*)alloc(128 * 160 * 2);
    _Float16* WoT  = (_Float16*)alloc(128 * 160 * 2);
    _Float16* Wc1T = (_Float16*)alloc(128 * 128 * 2);
    _Float16* F0T  = (_Float16*)alloc(128 * 544 * 2);
    _Float16* F1T  = (_Float16*)alloc(128 * 128 * 2);
    // overlapped region: [t2 | P1 | P2] reused as cat16 after edge kernel
    char* ovl = alloc((size_t)N * HDIM * 2 + 2 * (size_t)N * HDIM * 4);
    _Float16* t2 = (_Float16*)ovl;
    float* P1 = (float*)(ovl + alignUp((size_t)N * HDIM * 2, 256));
    float* P2 = P1 + (size_t)N * HDIM;
    _Float16* cat16 = (_Float16*)ovl;                 // N x 544, needs 1088*N bytes
    _Float16* f1 = t1;
    float* outF = (float*)d_out;
    (void)ws_size; (void)out_size;

    const int gBlk = (N + 127) / 128;

    // ---- prep ----
    init_kernel<<<2048, 256, 0, stream>>>(sumB, maxB, cntB, N);
    hf_kernel<<<(N + 7) / 8, 256, 0, stream>>>(x, hf16, N);
    wprep_kernel<<<(128 * 160 + 255) / 256, 256, 0, stream>>>(sW0, 129, 128, 128, 128, 160, W0sT);
    wprep_kernel<<<(128 * 128 + 255) / 256, 256, 0, stream>>>(sW1, 128, 128, 128, 128, 128, W1sT);
    wprep_kernel<<<(16 * 128 + 255) / 256, 256, 0, stream>>>(sW2, 128, 8, 8, 16, 128, W2sT);
    conv0_prep_kernel<<<(128 * 160 + 255) / 256, 256, 0, stream>>>(cW0, WeT, WoT);
    wprep_kernel<<<(128 * 128 + 255) / 256, 256, 0, stream>>>(cW1, 128, 128, 128, 128, 128, Wc1T);
    wprep_kernel<<<(128 * 544 + 255) / 256, 256, 0, stream>>>(fW0, 513, 128, 128, 128, 544, F0T);
    wprep_kernel<<<(128 * 128 + 255) / 256, 256, 0, stream>>>(fW1, 128, 128, 128, 128, 128, F1T);

    // ---- spatial MLP -> sf ----
    gemm128_kernel<<<gBlk, 256, 0, stream>>>(hf16, 160, W0sT, sb0, sg0, sbe0, N, 3, t1, nullptr);
    gemm128_kernel<<<gBlk, 256, 0, stream>>>(t1, 128, W1sT, sb1, sg1, sbe1, N, 3, t2, nullptr);
    gemm_emb_kernel<<<gBlk, 256, 0, stream>>>(t2, W2sT, sb2, N, sfB);

    // ---- node-side conv layer0 halves: P1 = hf@We, P2 = hf@Wo ----
    gemm128_kernel<<<gBlk, 256, 0, stream>>>(hf16, 160, WeT, nullptr, nullptr, nullptr, N, 0, nullptr, P1);
    gemm128_kernel<<<gBlk, 256, 0, stream>>>(hf16, 160, WoT, nullptr, nullptr, nullptr, N, 0, nullptr, P2);

    // ---- fused per-edge conv + gravity weight + scatter ----
    edge_kernel<<<(E + ET - 1) / ET, 128, 0, stream>>>(
        eidx, E, sfB, P1, P2, Wc1T, cb0, cg0, cbe0, cb1, sumB, maxB, cntB);

    // ---- aggregate + concat -> feature MLP -> out ----
    finalize_kernel<<<2048, 256, 0, stream>>>(sumB, maxB, cntB, hf16, cat16, N);
    gemm128_kernel<<<gBlk, 256, 0, stream>>>(cat16, 544, F0T, fb0, fg0, fbe0, N, 3, f1, nullptr);
    gemm128_kernel<<<gBlk, 256, 0, stream>>>(f1, 128, F1T, fb1, fg1, fbe1, N, 1, nullptr, outF);

    // tuple output: (out [N,128] f32, edge_index raw bits)
    (void)hipMemcpyAsync(outF + (size_t)N * HDIM, eidx, sizeof(int) * 2 * (size_t)E,
                         hipMemcpyDeviceToDevice, stream);
}
